// PG_50672024158610
// MI455X (gfx1250) — matrix operand order repaired and checked by reading, not compiled, not run on hardware
//
#include <hip/hip_runtime.h>

typedef unsigned short u16t;
typedef unsigned int   u32t;
typedef __attribute__((ext_vector_type(16))) __bf16 v16bf;
typedef __attribute__((ext_vector_type(8)))  float  v8f;
typedef __attribute__((ext_vector_type(4)))  u32t   u32x4;
typedef __attribute__((ext_vector_type(4)))  float  f32x4;

union Frag16 { v16bf v; u32x4 q[2]; u16t u[16]; };

__constant__ int c_branch[21] = {0,0,0,0,1,2,2,2,2,2,2,2,2,2,2,3,3,3,3,3,3};
__constant__ int c_pidx[21]   = {0,0,0,0,0,1,2,3,4,5,6,7,8,9,10,0,1,2,3,4,5};

#define ACT_ROWS 197            // 196 spatial rows + 1 zero row (row 196)
#define LDS_STRIDE 136          // 128 + 8 ushort pad -> 68-dword rows, bank skew 4

__device__ inline u16t f2bf(float f) {            // RNE fp32 -> bf16
  u32t x = __float_as_uint(f);
  x += 0x7FFFu + ((x >> 16) & 1u);
  return (u16t)(x >> 16);
}
__device__ inline float bf2f(u16t u) { return __uint_as_float(((u32t)u) << 16); }
__device__ inline u32t pack2(float a, float b) { return (u32t)f2bf(a) | ((u32t)f2bf(b) << 16); }

// A fragment: row-major [M][K]; lane row m fixed; e<8 -> k0+half*8+e ; e>=8 -> k0+16+half*8+(e-8)
__device__ __forceinline__ v16bf load_a(const u16t* arow, int k0, int half) {
  Frag16 f;
  f.q[0] = *(const u32x4*)(arow + k0 + half * 8);
  f.q[1] = *(const u32x4*)(arow + k0 + 16 + half * 8);
  return f.v;
}
// B fragment: [N][K] rows; lane col n fixed; e<8 -> k0+half*8+e ; e>=8 -> k0+16+half*8+(e-8)
__device__ __forceinline__ v16bf load_b(const u16t* brow, int k0, int half) {
  Frag16 f;
  f.q[0] = *(const u32x4*)(brow + k0 + half * 8);
  f.q[1] = *(const u32x4*)(brow + k0 + 16 + half * 8);
  return f.v;
}

#define WMMA_BF16(a, b, c) \
  __builtin_amdgcn_wmma_f32_16x16x32_bf16(false, (a), false, (b), (short)0, (c), false, false)

__device__ __forceinline__ v8f bias_acc(const float* bias, int mbase, int lane) {
  int half = lane >> 4;
  v8f c;
#pragma unroll
  for (int v = 0; v < 8; ++v) c[v] = bias[mbase + half * 8 + v];
  return c;
}

// shifted spatial row for tap (dy,dx); OOB -> zero row 196
__device__ __forceinline__ int shifted_row(int s, int dy, int dx) {
  int y = s / 14, x = s - y * 14;
  int yy = y + dy, xx = x + dx;
  bool ok = (s < 196) && ((unsigned)yy < 14u) && ((unsigned)xx < 14u);
  return ok ? (yy * 14 + xx) : 196;
}

// epilogue: (+optional residual [n][128] global) -> relu -> bf16 pack -> [n][Cst] store
__device__ __forceinline__ void epi_store(u16t* out, int Cst, const u16t* res,
                                          int mbase, int nbase, int lane, v8f acc) {
  int half = lane >> 4, nl = lane & 15;
  int n = nbase + nl;
  if (n >= 196) return;
  float v[8];
  if (res) {
    u32x4 rq = *(const u32x4*)(res + (size_t)n * 128 + mbase + half * 8);
    Frag16 rf; rf.q[0] = rq; rf.q[1] = rq;
#pragma unroll
    for (int j = 0; j < 8; ++j) v[j] = acc[j] + bf2f(rf.u[j]);
  } else {
#pragma unroll
    for (int j = 0; j < 8; ++j) v[j] = acc[j];
  }
#pragma unroll
  for (int j = 0; j < 8; ++j) v[j] = fmaxf(v[j], 0.0f);
  u32x4 q;
  q[0] = pack2(v[0], v[1]); q[1] = pack2(v[2], v[3]);
  q[2] = pack2(v[4], v[5]); q[3] = pack2(v[6], v[7]);
  *(u32x4*)(out + (size_t)n * Cst + mbase + half * 8) = q;
}

// cooperative copy: global act [197][128] -> LDS [197][136] (row 196 already zero in global)
__device__ __forceinline__ void stage_in(const u16t* g, u16t* sin) {
  for (int t = threadIdx.x; t < ACT_ROWS * 16; t += 256) {
    int row = t >> 4, c8 = t & 15;
    *(u32x4*)(sin + row * LDS_STRIDE + c8 * 8) =
        *(const u32x4*)(g + (size_t)row * 128 + c8 * 8);
  }
}

// conv3x3, Cin=128, B from LDS; each wave owns one m-tile, register-blocks 13 n-tiles
__device__ __forceinline__ void conv3x3_lds(const u16t* wb, const float* bias,
                                            const u16t* sin, u16t* out,
                                            const u16t* res, int wave, int lane) {
  int half = lane >> 4, nl = lane & 15;
  int mbase = wave * 16;
  v8f binit = bias_acc(bias, mbase, lane);
  v8f acc[13];
#pragma unroll
  for (int nt = 0; nt < 13; ++nt) acc[nt] = binit;
  const u16t* arow0 = wb + (size_t)(mbase + nl) * 128;
#pragma unroll 1
  for (int r = 0; r < 9; ++r) {
    int dy = r / 3 - 1, dx = r % 3 - 1;
    int sp[13];
#pragma unroll
    for (int nt = 0; nt < 13; ++nt) sp[nt] = shifted_row(nt * 16 + nl, dy, dx);
    const u16t* arowr = arow0 + (size_t)r * (128 * 128);
#pragma unroll 1
    for (int kc = 0; kc < 128; kc += 32) {
      v16bf a = load_a(arowr, kc, half);
#pragma unroll
      for (int nt = 0; nt < 13; ++nt) {
        v16bf bf = load_b(sin + (size_t)sp[nt] * LDS_STRIDE, kc, half);
        acc[nt] = WMMA_BF16(a, bf, acc[nt]);
      }
    }
  }
#pragma unroll
  for (int nt = 0; nt < 13; ++nt)
    epi_store(out, 128, res, mbase, nt * 16, lane, acc[nt]);
}

// stem conv1: Cin=1024, B from global [197][1024] (zero row 196)
__device__ __forceinline__ void conv3x3_g(const u16t* wb, const float* bias,
                                          const u16t* gin, u16t* out,
                                          int wave, int lane) {
  int half = lane >> 4, nl = lane & 15;
  int mbase = wave * 16;
  v8f binit = bias_acc(bias, mbase, lane);
  v8f acc[13];
#pragma unroll
  for (int nt = 0; nt < 13; ++nt) acc[nt] = binit;
  const u16t* arow0 = wb + (size_t)(mbase + nl) * 1024;
#pragma unroll 1
  for (int r = 0; r < 9; ++r) {
    int dy = r / 3 - 1, dx = r % 3 - 1;
    int sp[13];
#pragma unroll
    for (int nt = 0; nt < 13; ++nt) sp[nt] = shifted_row(nt * 16 + nl, dy, dx);
    const u16t* arowr = arow0 + (size_t)r * (128 * 1024);
#pragma unroll 1
    for (int kc = 0; kc < 1024; kc += 32) {
      v16bf a = load_a(arowr, kc, half);
#pragma unroll
      for (int nt = 0; nt < 13; ++nt) {
        v16bf bf = load_b(gin + (size_t)sp[nt] * 1024, kc, half);
        acc[nt] = WMMA_BF16(a, bf, acc[nt]);
      }
    }
  }
#pragma unroll
  for (int nt = 0; nt < 13; ++nt)
    epi_store(out, 128, nullptr, mbase, nt * 16, lane, acc[nt]);
}

// binary 1x1 proj: concat(a from LDS, b from global), K=256
__device__ __forceinline__ void proj_lds(const u16t* wb, const float* bias,
                                         const u16t* sin, const u16t* gb,
                                         u16t* out, int wave, int lane) {
  int half = lane >> 4, nl = lane & 15;
  int mbase = wave * 16;
  v8f binit = bias_acc(bias, mbase, lane);
  v8f acc[13];
  int srow[13];
#pragma unroll
  for (int nt = 0; nt < 13; ++nt) {
    acc[nt] = binit;
    int s = nt * 16 + nl;
    srow[nt] = (s < 196) ? s : 196;
  }
  const u16t* arow = wb + (size_t)(mbase + nl) * 256;
#pragma unroll 1
  for (int kc = 0; kc < 128; kc += 32) {
    v16bf a = load_a(arow, kc, half);
#pragma unroll
    for (int nt = 0; nt < 13; ++nt) {
      v16bf bf = load_b(sin + (size_t)srow[nt] * LDS_STRIDE, kc, half);
      acc[nt] = WMMA_BF16(a, bf, acc[nt]);
    }
  }
#pragma unroll 1
  for (int kc = 0; kc < 128; kc += 32) {
    v16bf a = load_a(arow + 128, kc, half);
#pragma unroll
    for (int nt = 0; nt < 13; ++nt) {
      v16bf bf = load_b(gb + (size_t)srow[nt] * 128, kc, half);
      acc[nt] = WMMA_BF16(a, bf, acc[nt]);
    }
  }
#pragma unroll
  for (int nt = 0; nt < 13; ++nt)
    epi_store(out, 128, nullptr, mbase, nt * 16, lane, acc[nt]);
}

// classifier 1x1: M=512 (4 m-tiles per wave), B from LDS, out [196][512]
__device__ __forceinline__ void cls_lds(const u16t* wb, const float* bias,
                                        const u16t* sin, u16t* out, int wave, int lane) {
  int half = lane >> 4, nl = lane & 15;
  int srow[13];
#pragma unroll
  for (int nt = 0; nt < 13; ++nt) {
    int s = nt * 16 + nl;
    srow[nt] = (s < 196) ? s : 196;
  }
#pragma unroll 1
  for (int mq = 0; mq < 4; ++mq) {
    int mbase = (mq * 8 + wave) * 16;
    v8f binit = bias_acc(bias, mbase, lane);
    v8f acc[13];
#pragma unroll
    for (int nt = 0; nt < 13; ++nt) acc[nt] = binit;
    const u16t* arow = wb + (size_t)(mbase + nl) * 128;
#pragma unroll 1
    for (int kc = 0; kc < 128; kc += 32) {
      v16bf a = load_a(arow, kc, half);
#pragma unroll
      for (int nt = 0; nt < 13; ++nt) {
        v16bf bf = load_b(sin + (size_t)srow[nt] * LDS_STRIDE, kc, half);
        acc[nt] = WMMA_BF16(a, bf, acc[nt]);
      }
    }
#pragma unroll
    for (int nt = 0; nt < 13; ++nt)
      epi_store(out, 512, nullptr, mbase, nt * 16, lane, acc[nt]);
  }
}

// ---- per-image megakernel: stem + program scan + classifier ----
struct NmnP {
  const int* programs;
  const float *sb1, *sb2, *ub1, *ub2, *pb, *b1, *b2, *clsb;
  const u16t *ws1, *ws2, *wu1, *wu2, *wp, *wb1, *wb2, *wcls;
  const u16t* featsb;   // [64][197][1024] bf16 (row 196 zero)
  u16t* acts;           // [64][6][197*128] bf16 (row 196 zero; bufs 0..4 scratch, 5 = feat)
  u16t* clsact;         // [64][196][512] bf16
};

#define WG_SYNC() do { __threadfence_block(); __syncthreads(); } while (0)

__global__ __launch_bounds__(256) void k_nmn(NmnP P) {
  __shared__ __align__(16) u16t s_in[ACT_ROWS * LDS_STRIDE];
  int b = blockIdx.x;
  int wave = threadIdx.x >> 5, lane = threadIdx.x & 31;
  u16t* bufs[6];
#pragma unroll
  for (int i = 0; i < 6; ++i) bufs[i] = P.acts + ((size_t)b * 6 + i) * (ACT_ROWS * 128);
  const u16t* fin = P.featsb + (size_t)b * ACT_ROWS * 1024;

  // stem
  conv3x3_g(P.ws1, P.sb1, fin, bufs[0], wave, lane);
  WG_SYNC();
  stage_in(bufs[0], s_in); WG_SYNC();
  conv3x3_lds(P.ws2, P.sb2, s_in, bufs[5], nullptr, wave, lane);
  WG_SYNC();

  // program scan (reverse), carry = (out, saved), both start at feat (id 5)
  int out_id = 5, saved_id = 5;
  int pool[5] = {0, 1, 2, 3, 4};
  int np = 5;
  for (int i = 24; i >= 0; --i) {
    int tok = P.programs[b * 25 + i];
    if (tok < 0 || tok > 20) tok = 0;
    int br = c_branch[tok], pi = c_pidx[tok];
    if (br == 1 || br == 2) {                       // scene / unary residual block
      int t1 = pool[--np], t2 = pool[--np];
      const u16t* x = bufs[(br == 1) ? 5 : out_id];
      stage_in(x, s_in); WG_SYNC();
      conv3x3_lds(P.wu1 + (size_t)pi * 9 * 128 * 128, P.ub1 + pi * 128,
                  s_in, bufs[t1], nullptr, wave, lane);
      WG_SYNC();
      stage_in(bufs[t1], s_in); WG_SYNC();
      conv3x3_lds(P.wu2 + (size_t)pi * 9 * 128 * 128, P.ub2 + pi * 128,
                  s_in, bufs[t2], x, wave, lane);
      WG_SYNC();
      pool[np++] = t1;
      if (br == 1) {
        if (saved_id != 5 && saved_id != out_id) pool[np++] = saved_id;
        saved_id = out_id;
        out_id = t2;
      } else {
        if (out_id != 5 && out_id != saved_id) pool[np++] = out_id;
        out_id = t2;
      }
    } else if (br == 3) {                           // binary: proj + residual block
      int t1 = pool[--np], t2 = pool[--np], t3 = pool[--np];
      stage_in(bufs[out_id], s_in); WG_SYNC();
      proj_lds(P.wp + (size_t)pi * 128 * 256, P.pb + pi * 128,
               s_in, bufs[saved_id], bufs[t1], wave, lane);
      WG_SYNC();
      stage_in(bufs[t1], s_in); WG_SYNC();
      conv3x3_lds(P.wb1 + (size_t)pi * 9 * 128 * 128, P.b1 + pi * 128,
                  s_in, bufs[t2], nullptr, wave, lane);
      WG_SYNC();
      stage_in(bufs[t2], s_in); WG_SYNC();
      conv3x3_lds(P.wb2 + (size_t)pi * 9 * 128 * 128, P.b2 + pi * 128,
                  s_in, bufs[t3], bufs[t1], wave, lane);
      WG_SYNC();
      pool[np++] = t1; pool[np++] = t2;
      if (out_id != 5 && out_id != saved_id) pool[np++] = out_id;
      out_id = t3;
    }
  }
  // classifier 1x1 conv -> relu
  stage_in(bufs[out_id], s_in); WG_SYNC();
  cls_lds(P.wcls, P.clsb, s_in, P.clsact + (size_t)b * 196 * 512, wave, lane);
}

// ---- maxpool 2x2 on [196][512] bf16 -> pooled [64][25088] (k = c*49 + py*7 + px) ----
__global__ void k_pool(const u16t* clsact, u16t* pooled) {
  long t = (long)blockIdx.x * blockDim.x + threadIdx.x;
  const long total = 64L * 512 * 49;
  if (t >= total) return;
  int b = (int)(t / (512 * 49));
  int r = (int)(t % (512 * 49));
  int c = r / 49, p = r % 49;
  int py = p / 7, px = p % 7;
  const u16t* base = clsact + (size_t)b * 196 * 512 + c;
  float best = -3.4e38f; u16t bu = 0;
#pragma unroll
  for (int dy = 0; dy < 2; ++dy)
#pragma unroll
    for (int dx = 0; dx < 2; ++dx) {
      int s = (2 * py + dy) * 14 + (2 * px + dx);
      u16t u = base[(size_t)s * 512];
      float f = bf2f(u);
      if (f > best) { best = f; bu = u; }
    }
  pooled[(size_t)b * 25088 + c * 49 + p] = bu;
}

__device__ __forceinline__ v8f gemm_acc(const u16t* arow, const u16t* brow,
                                        int K, int half, v8f acc) {
  for (int k0 = 0; k0 < K; k0 += 32) {
    if (k0 + 128 < K) __builtin_prefetch(arow + k0 + 128, 0, 1);
    v16bf a = load_a(arow, k0, half);
    v16bf b = load_b(brow, k0, half);
    acc = WMMA_BF16(a, b, acc);
  }
  return acc;
}

// ---- fc1: M=1024 K=25088 N=64, relu, bf16 out h[64][1024] ----
__global__ __launch_bounds__(256) void k_fc1(const u16t* w, const float* bias,
                                             const u16t* pooled, u16t* h) {
  int wave = threadIdx.x >> 5, lane = threadIdx.x & 31;
  int half = lane >> 4, nl = lane & 15;
  int tile = blockIdx.x * 8 + wave;              // 32 blocks * 8 waves = 256 tiles
  int mbase = (tile & 63) * 16, nbase = (tile >> 6) * 16;
  v8f acc = bias_acc(bias, mbase, lane);
  const u16t* arow = w      + (size_t)(mbase + nl) * 25088;
  const u16t* brow = pooled + (size_t)(nbase + nl) * 25088;
  acc = gemm_acc(arow, brow, 25088, half, acc);
  int n = nbase + nl;
  u32x4 q;
  q[0] = pack2(fmaxf(acc[0], 0.f), fmaxf(acc[1], 0.f));
  q[1] = pack2(fmaxf(acc[2], 0.f), fmaxf(acc[3], 0.f));
  q[2] = pack2(fmaxf(acc[4], 0.f), fmaxf(acc[5], 0.f));
  q[3] = pack2(fmaxf(acc[6], 0.f), fmaxf(acc[7], 0.f));
  *(u32x4*)(h + (size_t)n * 1024 + mbase + half * 8) = q;
}

// ---- fc2: M=32 K=1024 N=64, fp32 out [64][32] ----
__global__ __launch_bounds__(256) void k_fc2(const u16t* w, const float* bias,
                                             const u16t* h, float* out) {
  int wave = threadIdx.x >> 5, lane = threadIdx.x & 31;
  int half = lane >> 4, nl = lane & 15;
  int mbase = (wave & 1) * 16, nbase = (wave >> 1) * 16;   // 8 waves = 2x4 tiles
  v8f acc = bias_acc(bias, mbase, lane);
  const u16t* arow = w + (size_t)(mbase + nl) * 1024;
  const u16t* brow = h + (size_t)(nbase + nl) * 1024;
  acc = gemm_acc(arow, brow, 1024, half, acc);
  int n = nbase + nl;
  f32x4 f0, f1;
  f0[0] = acc[0]; f0[1] = acc[1]; f0[2] = acc[2]; f0[3] = acc[3];
  f1[0] = acc[4]; f1[1] = acc[5]; f1[2] = acc[6]; f1[3] = acc[7];
  float* o = out + (size_t)n * 32 + mbase + half * 8;
  *(f32x4*)o = f0;
  *(f32x4*)(o + 4) = f1;
}

// ---- repack kernels: fp32 -> bf16 banks ----
__global__ void k_wbank3x3(const float* src, u16t* dst, int G, int O, int I) {
  long total = (long)G * O * I * 9;
  for (long t = (long)blockIdx.x * blockDim.x + threadIdx.x; t < total;
       t += (long)gridDim.x * blockDim.x) {
    int r = (int)(t % 9);
    long t2 = t / 9;
    int i = (int)(t2 % I);
    long t3 = t2 / I;
    int o = (int)(t3 % O);
    int g = (int)(t3 / O);
    dst[(((size_t)g * 9 + r) * O + o) * I + i] = f2bf(src[t]);
  }
}
__global__ void k_flatbf(const float* src, u16t* dst, long n) {
  for (long t = (long)blockIdx.x * blockDim.x + threadIdx.x; t < n;
       t += (long)gridDim.x * blockDim.x)
    dst[t] = f2bf(src[t]);
}
// feats NCHW fp32 [64][1024][196] -> [64][197][1024] bf16 (spatial-major, zero row separate)
__global__ void k_featsb(const float* src, u16t* dst) {
  const long total = 64L * 1024 * 196;
  for (long t = (long)blockIdx.x * blockDim.x + threadIdx.x; t < total;
       t += (long)gridDim.x * blockDim.x) {
    int s = (int)(t % 196);
    long t2 = t / 196;
    int c = (int)(t2 % 1024);
    int b = (int)(t2 / 1024);
    dst[((size_t)b * ACT_ROWS + s) * 1024 + c] = f2bf(src[t]);
  }
}
// zero the guard rows (row 196) of all act buffers and featsb
__global__ void k_zrows(u16t* acts, u16t* featsb) {
  int t = blockIdx.x * blockDim.x + threadIdx.x;
  if (t < 64 * 6 * 128) {
    int buf = t >> 7, c = t & 127;
    acts[(size_t)buf * (ACT_ROWS * 128) + 196 * 128 + c] = 0;
  }
  if (t < 64 * 1024) {
    int bb = t >> 10, c = t & 1023;
    featsb[((size_t)bb * ACT_ROWS + 196) * 1024 + c] = 0;
  }
}

extern "C" void kernel_launch(void* const* d_in, const int* in_sizes, int n_in,
                              void* d_out, int out_size, void* d_ws, size_t ws_size,
                              hipStream_t stream) {
  (void)in_sizes; (void)n_in; (void)out_size; (void)ws_size;
  const float* feats    = (const float*)d_in[0];
  const int*   programs = (const int*)d_in[1];
  const float* stem_w1  = (const float*)d_in[2];
  const float* stem_b1  = (const float*)d_in[3];
  const float* stem_w2  = (const float*)d_in[4];
  const float* stem_b2  = (const float*)d_in[5];
  const float* uw1      = (const float*)d_in[6];
  const float* ub1      = (const float*)d_in[7];
  const float* uw2      = (const float*)d_in[8];
  const float* ub2      = (const float*)d_in[9];
  const float* bwp      = (const float*)d_in[10];
  const float* bbp      = (const float*)d_in[11];
  const float* bw1      = (const float*)d_in[12];
  const float* bb1      = (const float*)d_in[13];
  const float* bw2      = (const float*)d_in[14];
  const float* bb2      = (const float*)d_in[15];
  const float* cls_w    = (const float*)d_in[16];
  const float* cls_b    = (const float*)d_in[17];
  const float* fc1_w    = (const float*)d_in[18];
  const float* fc1_b    = (const float*)d_in[19];
  const float* fc2_w    = (const float*)d_in[20];
  const float* fc2_b    = (const float*)d_in[21];

  char* wsb = (char*)d_ws;
  size_t off = 0;
  auto alloc = [&](size_t bytes) -> void* {
    void* p = wsb + off;
    off += (bytes + 255) & ~(size_t)255;
    return p;
  };
  u16t* ws1b  = (u16t*)alloc(9UL * 128 * 1024 * 2);
  u16t* ws2b  = (u16t*)alloc(9UL * 128 * 128 * 2);
  u16t* wu1b  = (u16t*)alloc(11UL * 9 * 128 * 128 * 2);
  u16t* wu2b  = (u16t*)alloc(11UL * 9 * 128 * 128 * 2);
  u16t* wpb   = (u16t*)alloc(6UL * 128 * 256 * 2);
  u16t* wb1b  = (u16t*)alloc(6UL * 9 * 128 * 128 * 2);
  u16t* wb2b  = (u16t*)alloc(6UL * 9 * 128 * 128 * 2);
  u16t* wclsb = (u16t*)alloc(512UL * 128 * 2);
  u16t* wfc1b = (u16t*)alloc(1024UL * 25088 * 2);
  u16t* wfc2b = (u16t*)alloc(32UL * 1024 * 2);
  u16t* featsb = (u16t*)alloc(64UL * ACT_ROWS * 1024 * 2);
  u16t* acts   = (u16t*)alloc(64UL * 6 * ACT_ROWS * 128 * 2);
  u16t* clsact = (u16t*)alloc(64UL * 196 * 512 * 2);
  u16t* pooled = (u16t*)alloc(64UL * 25088 * 2);
  u16t* hbuf   = (u16t*)alloc(64UL * 1024 * 2);

  auto nb = [](long n) { return (unsigned)((n + 255) / 256); };

  k_zrows<<<256, 256, 0, stream>>>(acts, featsb);
  k_wbank3x3<<<nb(1L * 128 * 1024 * 9), 256, 0, stream>>>(stem_w1, ws1b, 1, 128, 1024);
  k_wbank3x3<<<nb(1L * 128 * 128 * 9),  256, 0, stream>>>(stem_w2, ws2b, 1, 128, 128);
  k_wbank3x3<<<nb(11L * 128 * 128 * 9), 256, 0, stream>>>(uw1, wu1b, 11, 128, 128);
  k_wbank3x3<<<nb(11L * 128 * 128 * 9), 256, 0, stream>>>(uw2, wu2b, 11, 128, 128);
  k_wbank3x3<<<nb(6L * 128 * 128 * 9),  256, 0, stream>>>(bw1, wb1b, 6, 128, 128);
  k_wbank3x3<<<nb(6L * 128 * 128 * 9),  256, 0, stream>>>(bw2, wb2b, 6, 128, 128);
  k_flatbf<<<nb(6L * 128 * 256),   256, 0, stream>>>(bwp,   wpb,   6L * 128 * 256);
  k_flatbf<<<nb(512L * 128),       256, 0, stream>>>(cls_w, wclsb, 512L * 128);
  k_flatbf<<<nb(1024L * 25088),    256, 0, stream>>>(fc1_w, wfc1b, 1024L * 25088);
  k_flatbf<<<nb(32L * 1024),       256, 0, stream>>>(fc2_w, wfc2b, 32L * 1024);
  k_featsb<<<nb(64L * 1024 * 196), 256, 0, stream>>>(feats, featsb);

  NmnP P;
  P.programs = programs;
  P.sb1 = stem_b1; P.sb2 = stem_b2;
  P.ub1 = ub1; P.ub2 = ub2; P.pb = bbp; P.b1 = bb1; P.b2 = bb2; P.clsb = cls_b;
  P.ws1 = ws1b; P.ws2 = ws2b; P.wu1 = wu1b; P.wu2 = wu2b;
  P.wp = wpb; P.wb1 = wb1b; P.wb2 = wb2b; P.wcls = wclsb;
  P.featsb = featsb; P.acts = acts; P.clsact = clsact;
  k_nmn<<<64, 256, 0, stream>>>(P);

  k_pool<<<nb(64L * 512 * 49), 256, 0, stream>>>(clsact, pooled);
  k_fc1<<<32, 256, 0, stream>>>(wfc1b, fc1_b, pooled, hbuf);
  k_fc2<<<1, 256, 0, stream>>>(wfc2b, fc2_b, hbuf, (float*)d_out);
}
